// PINN_70317204570379
// MI455X (gfx1250) — matrix operand order repaired and checked by reading, not compiled, not run on hardware
//
#include <hip/hip_runtime.h>
#include <math.h>

typedef __attribute__((ext_vector_type(16))) _Float16 v16h;
typedef __attribute__((ext_vector_type(8)))  _Float16 v8h;
typedef __attribute__((ext_vector_type(8)))  float    v8f;

#define NROWS 4096

// ---------------------------------------------------------------------------
// Fragment loaders (wave32 CDNA5 WMMA 16x16x32 f16 layouts, CDNA5 ISA)
// A (16x32, MxK): lane L<16 -> row L, K elems {0..7,16..23}; lane L>=16 -> row
// L-16, K elems {8..15,24..31}.  => two contiguous 16B loads at +hi*8 and +16.
// B (32x16, KxN): lane L<16 -> col L, K elems {0..7,16..23}; lane L>=16 -> col
// L-16, K elems {8..15,24..31}.  => two contiguous 16B loads at +hi*8 and +16.
// ---------------------------------------------------------------------------
__device__ __forceinline__ v16h frag_from2(v8h lo, v8h hi8) {
  v16h f;
#pragma unroll
  for (int i = 0; i < 8; ++i) { f[i] = lo[i]; f[i + 8] = hi8[i]; }
  return f;
}
__device__ __forceinline__ v16h load_a_frag(const _Float16* p) {
  return frag_from2(*(const v8h*)p, *(const v8h*)(p + 16));
}
__device__ __forceinline__ v16h load_b_frag(const _Float16* p) {
  return frag_from2(*(const v8h*)p, *(const v8h*)(p + 16));
}

// ---------------------------------------------------------------------------
// NT GEMM: C[m,n] = alpha * sum_k A[m,k]*B[n,k]  (+ bias[n]) (+ res[m,n]) (ReLU)
// A: MxK f16 (lda), B: NxK f16 (ldb), C: MxN f32 (ldc). M%16==0, N%16==0, K%32==0.
// Wave tile = (TM*16) x (TN*16); block = WM x WN waves (32*WM*WN threads).
//   <4,4,2,4>: 64x64 per wave, 128x256 per block  -> 16 WMMA per 16 b128 loads
//   <2,2,4,2>: 32x32 per wave, 128x64  per block  -> skinny projections
// ---------------------------------------------------------------------------
template <int TM, int TN, int WM, int WN>
__global__ void __launch_bounds__(32 * WM * WN)
gemm_nt_wmma(const _Float16* __restrict__ A, int lda,
             const _Float16* __restrict__ B, int ldb,
             float* __restrict__ C, int ldc,
             const float* __restrict__ bias,
             const float* __restrict__ res, int ldr,
             int M, int N, int K, float alpha, int relu)
{
  const int wave = threadIdx.x >> 5;
  const int lane = threadIdx.x & 31;
  const int wm = wave % WM, wn = wave / WM;
  const int m0 = blockIdx.x * (WM * TM * 16) + wm * (TM * 16);
  const int n0 = blockIdx.y * (WN * TN * 16) + wn * (TN * 16);
  if (m0 >= M || n0 >= N) return;
  const int r  = lane & 15;
  const int hi = lane >> 4;

  const v8f zero = {};
  v8f acc[TM][TN];
#pragma unroll
  for (int i = 0; i < TM; ++i)
#pragma unroll
    for (int j = 0; j < TN; ++j) acc[i][j] = zero;

  const _Float16* ap[TM];
  const _Float16* bp[TN];
#pragma unroll
  for (int i = 0; i < TM; ++i) ap[i] = A + (size_t)(m0 + i * 16 + r) * lda + hi * 8;
#pragma unroll
  for (int j = 0; j < TN; ++j) bp[j] = B + (size_t)(n0 + j * 16 + r) * ldb + hi * 8;

#pragma unroll 2
  for (int k = 0; k < K; k += 32) {
    v16h af[TM], bf[TN];
#pragma unroll
    for (int i = 0; i < TM; ++i) af[i] = load_a_frag(ap[i] + k);
#pragma unroll
    for (int j = 0; j < TN; ++j) bf[j] = load_b_frag(bp[j] + k);
#pragma unroll
    for (int i = 0; i < TM; ++i)
#pragma unroll
      for (int j = 0; j < TN; ++j)
        acc[i][j] = __builtin_amdgcn_wmma_f32_16x16x32_f16(
            false, af[i], false, bf[j], (short)0, acc[i][j], false, false);
  }

  // C/D layout: VGPR rr -> row = tileRow + hi*8 + rr ; col = tileCol + (lane&15)
  const int c = lane & 15;
#pragma unroll
  for (int tn = 0; tn < TN; ++tn) {
    const int col = n0 + tn * 16 + c;
    const float bb = bias ? bias[col] : 0.0f;
#pragma unroll
    for (int tm = 0; tm < TM; ++tm) {
      const int row = m0 + tm * 16 + hi * 8;
      float* outp = C + (size_t)row * ldc + col;
      const float* rp = res ? (res + (size_t)row * ldr + col) : nullptr;
#pragma unroll
      for (int rr = 0; rr < 8; ++rr) {
        float v = acc[tm][tn][rr] * alpha + bb;
        if (res)  v += rp[(size_t)rr * ldr];
        if (relu) v = fmaxf(v, 0.0f);
        outp[(size_t)rr * ldc] = v;
      }
    }
  }
}

// ---------------------------------------------------------------------------
// f32 -> f16 with zero padding (also plain conversion when dims match)
// ---------------------------------------------------------------------------
__global__ void f32_to_f16_pad(_Float16* __restrict__ dst, const float* __restrict__ src,
                               int dstRows, int dstCols, int srcRows, int srcCols)
{
  int i = blockIdx.x * blockDim.x + threadIdx.x;
  int total = dstRows * dstCols;
  if (i >= total) return;
  int rrow = i / dstCols, col = i % dstCols;
  float v = (rrow < srcRows && col < srcCols) ? src[(size_t)rrow * srcCols + col] : 0.0f;
  dst[i] = (_Float16)v;
}

// dst[n][k] = src[k][srcColOff + n]  (f32 -> f16); dst is rowsOut x colsOut
__global__ void transpose_f32_to_f16(_Float16* __restrict__ dst, const float* __restrict__ src,
                                     int srcLd, int srcColOff, int rowsOut, int colsOut)
{
  int i = blockIdx.x * blockDim.x + threadIdx.x;
  int total = rowsOut * colsOut;
  if (i >= total) return;
  int nn = i / colsOut, kk = i % colsOut;
  dst[i] = (_Float16)src[(size_t)kk * srcLd + srcColOff + nn];
}

// Pack hpm_inw (291x97) into (384x128): q rows 0..96, k rows 128..224, v rows 256..352
__global__ void pack_hpm_qkv_w(_Float16* __restrict__ dst, const float* __restrict__ src)
{
  int i = blockIdx.x * blockDim.x + threadIdx.x;
  if (i >= 384 * 128) return;
  int rrow = i / 128, col = i % 128;
  int blk = rrow / 128, sub = rrow % 128;
  float v = 0.0f;
  if (sub < 97 && col < 97) v = src[(size_t)(blk * 97 + sub) * 97 + col];
  dst[i] = (_Float16)v;
}
__global__ void pack_hpm_qkv_b(float* __restrict__ dst, const float* __restrict__ src)
{
  int i = blockIdx.x * blockDim.x + threadIdx.x;
  if (i >= 384) return;
  int blk = i / 128, sub = i % 128;
  dst[i] = (sub < 97) ? src[blk * 97 + sub] : 0.0f;
}
__global__ void pad_bias_f32(float* __restrict__ dst, const float* __restrict__ src,
                             int n, int valid)
{
  int i = blockIdx.x * blockDim.x + threadIdx.x;
  if (i >= n) return;
  dst[i] = (i < valid) ? src[i] : 0.0f;
}

// ---------------------------------------------------------------------------
// Row softmax (in place), one block per row
// ---------------------------------------------------------------------------
__global__ void __launch_bounds__(256)
softmax_rows(float* __restrict__ S, int ncols)
{
  __shared__ float red[256];
  const int row = blockIdx.x;
  float* p = S + (size_t)row * ncols;
  float m = -3.4e38f;
  for (int c = threadIdx.x; c < ncols; c += blockDim.x) m = fmaxf(m, p[c]);
  red[threadIdx.x] = m; __syncthreads();
  for (int st = 128; st > 0; st >>= 1) {
    if (threadIdx.x < st) red[threadIdx.x] = fmaxf(red[threadIdx.x], red[threadIdx.x + st]);
    __syncthreads();
  }
  m = red[0]; __syncthreads();
  float s = 0.0f;
  for (int c = threadIdx.x; c < ncols; c += blockDim.x) {
    float e = __expf(p[c] - m);
    p[c] = e; s += e;
  }
  red[threadIdx.x] = s; __syncthreads();
  for (int st = 128; st > 0; st >>= 1) {
    if (threadIdx.x < st) red[threadIdx.x] += red[threadIdx.x + st];
    __syncthreads();
  }
  const float inv = 1.0f / red[0];
  for (int c = threadIdx.x; c < ncols; c += blockDim.x) p[c] *= inv;
}

// ---------------------------------------------------------------------------
// LayerNorm over first `valid` columns; pad columns written as 0.
// ---------------------------------------------------------------------------
__global__ void __launch_bounds__(128)
layernorm_rows(float* __restrict__ dst, const float* __restrict__ src,
               const float* __restrict__ g, const float* __restrict__ b,
               int ld, int valid)
{
  __shared__ float red[128];
  const int row = blockIdx.x;
  const float* p = src + (size_t)row * ld;
  float* o = dst + (size_t)row * ld;
  float s = 0.0f;
  for (int c = threadIdx.x; c < valid; c += blockDim.x) s += p[c];
  red[threadIdx.x] = s; __syncthreads();
  for (int st = 64; st > 0; st >>= 1) {
    if (threadIdx.x < st) red[threadIdx.x] += red[threadIdx.x + st];
    __syncthreads();
  }
  const float mean = red[0] / (float)valid;
  __syncthreads();
  float v = 0.0f;
  for (int c = threadIdx.x; c < valid; c += blockDim.x) { float d = p[c] - mean; v += d * d; }
  red[threadIdx.x] = v; __syncthreads();
  for (int st = 64; st > 0; st >>= 1) {
    if (threadIdx.x < st) red[threadIdx.x] += red[threadIdx.x + st];
    __syncthreads();
  }
  const float inv = rsqrtf(red[0] / (float)valid + 1e-5f);
  for (int c = threadIdx.x; c < ld; c += blockDim.x)
    o[c] = (c < valid) ? (p[c] - mean) * inv * g[c] + b[c] : 0.0f;
}

// ---------------------------------------------------------------------------
// Per-row tanh-MLP: u, du/dt, g1 = du/dh, g2 = d(sum g1)/dh = H u . 1
// (forward-over-reverse dual backprop). Assembles deri = [h | u | g1 | g2 | 0pad]
// ---------------------------------------------------------------------------
__global__ void __launch_bounds__(256)
rowmlp_kernel(const float* __restrict__ h, const int* __restrict__ idx,
              const float* __restrict__ w0, const float* __restrict__ b0,
              const float* __restrict__ wm, const float* __restrict__ bm,
              const float* __restrict__ w5, const float* __restrict__ b5,
              const float* __restrict__ pp,
              float* __restrict__ u_out, float* __restrict__ ut_out,
              float* __restrict__ deri, _Float16* __restrict__ deri_h,
              int nrows)
{
  __shared__ float sw0[330], sb0[10], swm[400], sbm[40], sr5[10], sconst[1];
  const int t = threadIdx.x;
  for (int i = t; i < 330; i += blockDim.x) sw0[i] = w0[i];
  for (int i = t; i < 400; i += blockDim.x) swm[i] = wm[i];
  for (int i = t; i < 40;  i += blockDim.x) sbm[i] = bm[i];
  if (t < 10) {
    sb0[t] = b0[t];
    float s = 0.0f;
    for (int j = 0; j < 6; ++j) s += w5[j * 10 + t] * pp[j];
    sr5[t] = s;                       // r5 = w5^T p  (du/dz5, constant)
  }
  if (t == 0) {
    float s = 0.0f;
    for (int j = 0; j < 6; ++j) s += pp[j] * b5[j];
    sconst[0] = s;                    // p . b5
  }
  __syncthreads();
  const int row = blockIdx.x * blockDim.x + t;
  if (row >= nrows) return;

  float z0[33];
#pragma unroll
  for (int i = 0; i < 32; ++i) z0[i] = h[(size_t)row * 32 + i];
  z0[32] = (float)idx[row];

  // forward + tangent (tangent dir = ones on h, 0 on t)
  float z[5][10], zd[5][10];
  for (int i = 0; i < 10; ++i) {
    float a = sb0[i], ad = 0.0f;
    const float* W = &sw0[i * 33];
    for (int j = 0; j < 33; ++j) a += W[j] * z0[j];
    for (int j = 0; j < 32; ++j) ad += W[j];
    float zz = tanhf(a);
    z[0][i] = zz;
    zd[0][i] = (1.0f - zz * zz) * ad;
  }
  for (int L = 1; L < 5; ++L) {
    const float* W = &swm[(L - 1) * 100];
    const float* bb = &sbm[(L - 1) * 10];
    float nz[10], nzd[10];
    for (int i = 0; i < 10; ++i) {
      float a = bb[i], ad = 0.0f;
      for (int j = 0; j < 10; ++j) {
        a  += W[i * 10 + j] * z[L - 1][j];
        ad += W[i * 10 + j] * zd[L - 1][j];
      }
      float zz = tanhf(a);
      nz[i] = zz; nzd[i] = (1.0f - zz * zz) * ad;
    }
    for (int i = 0; i < 10; ++i) { z[L][i] = nz[i]; zd[L][i] = nzd[i]; }
  }
  float u = sconst[0];
  for (int i = 0; i < 10; ++i) u += sr5[i] * z[4][i];

  // reverse + dual reverse
  float r[10], rd[10];
  for (int i = 0; i < 10; ++i) { r[i] = sr5[i]; rd[i] = 0.0f; }
  float g[33], gd[33];
  for (int L = 4; L >= 0; --L) {
    float d[10], dd[10];
    for (int i = 0; i < 10; ++i) {
      float zz = z[L][i];
      float tt = 1.0f - zz * zz;
      d[i]  = r[i] * tt;
      dd[i] = rd[i] * tt - 2.0f * r[i] * zz * zd[L][i];
    }
    if (L > 0) {
      const float* W = &swm[(L - 1) * 100];
      float nr[10], nrd[10];
      for (int j = 0; j < 10; ++j) {
        float s = 0.0f, sd = 0.0f;
        for (int i = 0; i < 10; ++i) { s += W[i * 10 + j] * d[i]; sd += W[i * 10 + j] * dd[i]; }
        nr[j] = s; nrd[j] = sd;
      }
      for (int j = 0; j < 10; ++j) { r[j] = nr[j]; rd[j] = nrd[j]; }
    } else {
      for (int j = 0; j < 33; ++j) {
        float s = 0.0f, sd = 0.0f;
        for (int i = 0; i < 10; ++i) { s += sw0[i * 33 + j] * d[i]; sd += sw0[i * 33 + j] * dd[i]; }
        g[j] = s; gd[j] = sd;
      }
    }
  }

  u_out[row] = u;
  ut_out[row] = g[32];
  float drow[128];
  for (int i = 0; i < 32; ++i) drow[i] = z0[i];
  drow[32] = u;
  for (int i = 0; i < 32; ++i) drow[33 + i] = g[i];
  for (int i = 0; i < 32; ++i) drow[65 + i] = gd[i];
  for (int i = 97; i < 128; ++i) drow[i] = 0.0f;
  for (int i = 0; i < 128; ++i) {
    deri[(size_t)row * 128 + i] = drow[i];
    deri_h[(size_t)row * 128 + i] = (_Float16)drow[i];
  }
}

// f[row] = ut[row] - (hpm_d2b + sum_{c<97} d1out2[row][c] * hpm_d2w[c])
__global__ void final_f_kernel(const float* __restrict__ d1out2, int ld,
                               const float* __restrict__ d2w, const float* __restrict__ d2b,
                               const float* __restrict__ ut, float* __restrict__ f_out,
                               int nrows, int valid)
{
  int row = blockIdx.x * blockDim.x + threadIdx.x;
  if (row >= nrows) return;
  float s = d2b[0];
  const float* p = d1out2 + (size_t)row * ld;
  for (int c = 0; c < valid; ++c) s += p[c] * d2w[c];
  f_out[row] = ut[row] - s;
}

// ---------------------------------------------------------------------------
// Host side
// ---------------------------------------------------------------------------
extern "C" void kernel_launch(void* const* d_in, const int* in_sizes, int n_in,
                              void* d_out, int out_size, void* d_ws, size_t ws_size,
                              hipStream_t stream) {
  (void)in_sizes; (void)n_in; (void)out_size; (void)ws_size;
  const float* x    = (const float*)d_in[0];
  const int*   idx  = (const int*)d_in[1];
  const float* inw  = (const float*)d_in[2];
  const float* inb  = (const float*)d_in[3];
  const float* outw = (const float*)d_in[4];
  const float* outb = (const float*)d_in[5];
  const float* lng  = (const float*)d_in[6];
  const float* lnb  = (const float*)d_in[7];
  const float* d1w  = (const float*)d_in[8];
  const float* d1b  = (const float*)d_in[9];
  const float* d2w  = (const float*)d_in[10];
  const float* d2b  = (const float*)d_in[11];
  const float* mw0  = (const float*)d_in[12];
  const float* mb0  = (const float*)d_in[13];
  const float* mwm  = (const float*)d_in[14];
  const float* mbm  = (const float*)d_in[15];
  const float* mw5  = (const float*)d_in[16];
  const float* mb5  = (const float*)d_in[17];
  const float* mp   = (const float*)d_in[18];
  const float* hinw = (const float*)d_in[19];
  const float* hinb = (const float*)d_in[20];
  const float* houtw= (const float*)d_in[21];
  const float* houtb= (const float*)d_in[22];
  const float* hlng = (const float*)d_in[23];
  const float* hlnb = (const float*)d_in[24];
  const float* hd1w = (const float*)d_in[25];
  const float* hd1b = (const float*)d_in[26];
  const float* hd2w = (const float*)d_in[27];
  const float* hd2b = (const float*)d_in[28];

  float* out   = (float*)d_out;
  float* u_out = out;                         // (4096,)
  float* h_out = out + NROWS;                 // (4096,32)
  float* f_out = out + NROWS + NROWS * 32;    // (4096,1)

  // bump allocator over d_ws
  char* ws = (char*)d_ws;
  size_t off = 0;
  auto alloc = [&](size_t bytes) -> void* {
    void* p = ws + off;
    off = (off + bytes + 255) & ~(size_t)255;
    return p;
  };
  // f32 scratch
  float* S       = (float*)alloc((size_t)NROWS * NROWS * 4);
  float* qkv     = (float*)alloc((size_t)NROWS * 192 * 4);
  float* attn    = (float*)alloc((size_t)NROWS * 64 * 4);
  float* attnout = (float*)alloc((size_t)NROWS * 64 * 4);
  float* lnx     = (float*)alloc((size_t)NROWS * 64 * 4);
  float* d1o     = (float*)alloc((size_t)NROWS * 64 * 4);
  float* qkvp    = (float*)alloc((size_t)NROWS * 384 * 4);
  float* attn2   = (float*)alloc((size_t)NROWS * 128 * 4);
  float* attnout2= (float*)alloc((size_t)NROWS * 128 * 4);
  float* ln2     = (float*)alloc((size_t)NROWS * 128 * 4);
  float* d1o2    = (float*)alloc((size_t)NROWS * 128 * 4);
  float* deri    = (float*)alloc((size_t)NROWS * 128 * 4);
  float* ut      = (float*)alloc((size_t)NROWS * 4);
  float* b_hinb  = (float*)alloc(384 * 4);
  float* b_houtb = (float*)alloc(128 * 4);
  float* b_hd1b  = (float*)alloc(128 * 4);
  // f16 scratch
  _Float16* Sh      = (_Float16*)alloc((size_t)NROWS * NROWS * 2);
  _Float16* xh      = (_Float16*)alloc((size_t)NROWS * 64 * 2);
  _Float16* qkvh    = (_Float16*)alloc((size_t)NROWS * 192 * 2);
  _Float16* attnh   = (_Float16*)alloc((size_t)NROWS * 64 * 2);
  _Float16* lnh     = (_Float16*)alloc((size_t)NROWS * 64 * 2);
  _Float16* d1h     = (_Float16*)alloc((size_t)NROWS * 64 * 2);
  _Float16* vT      = (_Float16*)alloc((size_t)64 * NROWS * 2);
  _Float16* derih   = (_Float16*)alloc((size_t)NROWS * 128 * 2);
  _Float16* qkvph   = (_Float16*)alloc((size_t)NROWS * 384 * 2);
  _Float16* attn2h  = (_Float16*)alloc((size_t)NROWS * 128 * 2);
  _Float16* ln2h    = (_Float16*)alloc((size_t)NROWS * 128 * 2);
  _Float16* vT2     = (_Float16*)alloc((size_t)128 * NROWS * 2);
  _Float16* w_inw   = (_Float16*)alloc((size_t)192 * 64 * 2);
  _Float16* w_outw  = (_Float16*)alloc((size_t)64 * 64 * 2);
  _Float16* w_d1w   = (_Float16*)alloc((size_t)64 * 64 * 2);
  _Float16* w_d2w   = (_Float16*)alloc((size_t)32 * 64 * 2);
  _Float16* w_hinw  = (_Float16*)alloc((size_t)384 * 128 * 2);
  _Float16* w_houtw = (_Float16*)alloc((size_t)128 * 128 * 2);
  _Float16* w_hd1w  = (_Float16*)alloc((size_t)128 * 128 * 2);

  auto g1d = [](int total) { return dim3((unsigned)((total + 255) / 256)); };
  // small config: block tile 128 x 64 ; big config: block tile 128 x 256
  auto gemmGs = [](int M, int N) { return dim3((unsigned)((M + 127) / 128), (unsigned)((N + 63) / 64)); };
  auto gemmGb = [](int M, int N) { return dim3((unsigned)((M + 127) / 128), (unsigned)((N + 255) / 256)); };
  const dim3 B256(256), B128(128);
  const float sc1 = 1.0f / 8.0f;              // 1/sqrt(64)
  const float sc2 = 1.0f / sqrtf(97.0f);      // 1/sqrt(97)

  // ---- weight / input conversion ----
  f32_to_f16_pad<<<g1d(NROWS*64), B256, 0, stream>>>(xh, x, NROWS, 64, NROWS, 64);
  f32_to_f16_pad<<<g1d(192*64),  B256, 0, stream>>>(w_inw,  inw,  192, 64, 192, 64);
  f32_to_f16_pad<<<g1d(64*64),   B256, 0, stream>>>(w_outw, outw, 64,  64, 64,  64);
  f32_to_f16_pad<<<g1d(64*64),   B256, 0, stream>>>(w_d1w,  d1w,  64,  64, 64,  64);
  f32_to_f16_pad<<<g1d(32*64),   B256, 0, stream>>>(w_d2w,  d2w,  32,  64, 32,  64);
  pack_hpm_qkv_w<<<g1d(384*128), B256, 0, stream>>>(w_hinw, hinw);
  pack_hpm_qkv_b<<<g1d(384),     B256, 0, stream>>>(b_hinb, hinb);
  f32_to_f16_pad<<<g1d(128*128), B256, 0, stream>>>(w_houtw, houtw, 128, 128, 97, 97);
  pad_bias_f32<<<g1d(128),       B256, 0, stream>>>(b_houtb, houtb, 128, 97);
  f32_to_f16_pad<<<g1d(128*128), B256, 0, stream>>>(w_hd1w, hd1w, 128, 128, 97, 97);
  pad_bias_f32<<<g1d(128),       B256, 0, stream>>>(b_hd1b, hd1b, 128, 97);

  // ---- xnn block ----
  gemm_nt_wmma<2,2,4,2><<<gemmGs(NROWS,192), B256, 0, stream>>>(xh, 64, w_inw, 64, qkv, 192,
      inb, nullptr, 0, NROWS, 192, 64, 1.0f, 0);
  f32_to_f16_pad<<<g1d(NROWS*192), B256, 0, stream>>>(qkvh, qkv, NROWS, 192, NROWS, 192);
  gemm_nt_wmma<4,4,2,4><<<gemmGb(NROWS,NROWS), B256, 0, stream>>>(qkvh, 192, qkvh + 64, 192, S, NROWS,
      nullptr, nullptr, 0, NROWS, NROWS, 64, sc1, 0);
  softmax_rows<<<NROWS, B256, 0, stream>>>(S, NROWS);
  f32_to_f16_pad<<<g1d(NROWS*NROWS), B256, 0, stream>>>(Sh, S, NROWS, NROWS, NROWS, NROWS);
  transpose_f32_to_f16<<<g1d(64*NROWS), B256, 0, stream>>>(vT, qkv, 192, 128, 64, NROWS);
  gemm_nt_wmma<2,2,4,2><<<gemmGs(NROWS,64), B256, 0, stream>>>(Sh, NROWS, vT, NROWS, attn, 64,
      nullptr, nullptr, 0, NROWS, 64, NROWS, 1.0f, 0);
  f32_to_f16_pad<<<g1d(NROWS*64), B256, 0, stream>>>(attnh, attn, NROWS, 64, NROWS, 64);
  gemm_nt_wmma<2,2,4,2><<<gemmGs(NROWS,64), B256, 0, stream>>>(attnh, 64, w_outw, 64, attnout, 64,
      outb, x, 64, NROWS, 64, 64, 1.0f, 0);
  layernorm_rows<<<NROWS, B128, 0, stream>>>(lnx, attnout, lng, lnb, 64, 64);
  f32_to_f16_pad<<<g1d(NROWS*64), B256, 0, stream>>>(lnh, lnx, NROWS, 64, NROWS, 64);
  gemm_nt_wmma<2,2,4,2><<<gemmGs(NROWS,64), B256, 0, stream>>>(lnh, 64, w_d1w, 64, d1o, 64,
      d1b, lnx, 64, NROWS, 64, 64, 1.0f, 1);
  f32_to_f16_pad<<<g1d(NROWS*64), B256, 0, stream>>>(d1h, d1o, NROWS, 64, NROWS, 64);
  gemm_nt_wmma<2,2,4,2><<<gemmGs(NROWS,32), B256, 0, stream>>>(d1h, 64, w_d2w, 64, h_out, 32,
      d2b, nullptr, 0, NROWS, 32, 64, 1.0f, 0);

  // ---- per-row autodiff MLP ----
  rowmlp_kernel<<<g1d(NROWS), B256, 0, stream>>>(h_out, idx, mw0, mb0, mwm, mbm,
      mw5, mb5, mp, u_out, ut, deri, derih, NROWS);

  // ---- hpm block (D=97 padded to 128) ----
  gemm_nt_wmma<2,2,4,2><<<gemmGs(NROWS,384), B256, 0, stream>>>(derih, 128, w_hinw, 128, qkvp, 384,
      b_hinb, nullptr, 0, NROWS, 384, 128, 1.0f, 0);
  f32_to_f16_pad<<<g1d(NROWS*384), B256, 0, stream>>>(qkvph, qkvp, NROWS, 384, NROWS, 384);
  gemm_nt_wmma<4,4,2,4><<<gemmGb(NROWS,NROWS), B256, 0, stream>>>(qkvph, 384, qkvph + 128, 384, S, NROWS,
      nullptr, nullptr, 0, NROWS, NROWS, 128, sc2, 0);
  softmax_rows<<<NROWS, B256, 0, stream>>>(S, NROWS);
  f32_to_f16_pad<<<g1d(NROWS*NROWS), B256, 0, stream>>>(Sh, S, NROWS, NROWS, NROWS, NROWS);
  transpose_f32_to_f16<<<g1d(128*NROWS), B256, 0, stream>>>(vT2, qkvp, 384, 256, 128, NROWS);
  gemm_nt_wmma<2,2,4,2><<<gemmGs(NROWS,128), B256, 0, stream>>>(Sh, NROWS, vT2, NROWS, attn2, 128,
      nullptr, nullptr, 0, NROWS, 128, NROWS, 1.0f, 0);
  f32_to_f16_pad<<<g1d(NROWS*128), B256, 0, stream>>>(attn2h, attn2, NROWS, 128, NROWS, 128);
  gemm_nt_wmma<2,2,4,2><<<gemmGs(NROWS,128), B256, 0, stream>>>(attn2h, 128, w_houtw, 128, attnout2, 128,
      b_houtb, deri, 128, NROWS, 128, 128, 1.0f, 0);
  layernorm_rows<<<NROWS, B128, 0, stream>>>(ln2, attnout2, hlng, hlnb, 128, 97);
  f32_to_f16_pad<<<g1d(NROWS*128), B256, 0, stream>>>(ln2h, ln2, NROWS, 128, NROWS, 128);
  gemm_nt_wmma<2,2,4,2><<<gemmGs(NROWS,128), B256, 0, stream>>>(ln2h, 128, w_hd1w, 128, d1o2, 128,
      b_hd1b, ln2, 128, NROWS, 128, 128, 1.0f, 1);
  final_f_kernel<<<g1d(NROWS), B256, 0, stream>>>(d1o2, 128, hd2w, hd2b, ut, f_out, NROWS, 97);
}
